// CNILUT_37864431682406
// MI455X (gfx1250) — matrix operand order repaired and checked by reading, not compiled, not run on hardware
//
#include <hip/hip_runtime.h>

// ---------------------------------------------------------------------------
// CNILUT on MI455X (gfx1250): per-pixel MLP 6->256->(256x3, tanh)->3.
// Compute-bound (~0.42 TFLOP vs ~25MB traffic) -> f16 WMMA w/ f32 accum.
// One wave owns a 16-pixel tile; hidden layers run TWO independent
// v_wmma_f32_16x16x32_f16 accumulation chains (N-tiles nt and nt+8) off
// shared A registers for XDL pipelining; weights pre-transposed to f16.
// ---------------------------------------------------------------------------

typedef _Float16 h16;
typedef __attribute__((ext_vector_type(16))) _Float16 v16h;
typedef __attribute__((ext_vector_type(8)))  _Float16 v8h;
typedef __attribute__((ext_vector_type(8)))  float    v8f;

#define HW      262144          // 512*512
#define NPIX    1048576         // 4*512*512
#define NF      256
#define WAVES   4
#define PIX_PER_WG (WAVES * 16)
#define STRIDE  264             // halves per activation row (528B = 33*16B, bank-spread + b128-aligned)

// workspace byte offsets
#define WS_WT   0               // 3 * 256*256 f16 transposed weights (WT[l][n][k])
#define WS_C0   393216          // 256 f32: b0 + style @ W0[3:6]
#define WS_W4T  394240          // 3*256 f32: W4 transposed [c][k]

// ---------------------------------------------------------------------------
// Prep: f16-transpose hidden weights, fold style into layer-0 constant,
// transpose W4. Tiny (one-off, ~200K elements).
// ---------------------------------------------------------------------------
__global__ void cnilut_prep(const float* __restrict__ W0, const float* __restrict__ b0,
                            const float* __restrict__ W1, const float* __restrict__ W2,
                            const float* __restrict__ W3, const float* __restrict__ W4,
                            const float* __restrict__ style,
                            h16* __restrict__ wt, float* __restrict__ c0,
                            float* __restrict__ w4t) {
    int idx = blockIdx.x * 256 + threadIdx.x;
    if (idx < 3 * 65536) {
        int l = idx >> 16, r = idx & 65535;
        int n = r >> 8, k = r & 255;
        const float* W = (l == 0) ? W1 : (l == 1) ? W2 : W3;
        wt[l * 65536 + n * 256 + k] = (h16)W[k * 256 + n];   // WT[n][k] = W[k][n]
    } else if (idx < 3 * 65536 + 256) {
        int n = idx - 3 * 65536;
        c0[n] = b0[n] + style[0] * W0[3 * 256 + n]
                      + style[1] * W0[4 * 256 + n]
                      + style[2] * W0[5 * 256 + n];
    } else if (idx < 3 * 65536 + 256 + 768) {
        int r = idx - (3 * 65536 + 256);
        int c = r >> 8, k = r & 255;
        w4t[c * 256 + k] = W4[k * 3 + c];
    }
}

__device__ __forceinline__ float fast_tanh(float v) {
    // tanh(x) = 1 - 2/(exp2(2x*log2e)+1); v_exp2/v_rcp are TRANS ops that
    // co-execute with XDL WMMA on CDNA5.
    float e = __builtin_amdgcn_exp2f(v * 2.885390081777927f);
    return 1.0f - 2.0f * __builtin_amdgcn_rcpf(e + 1.0f);
}

// ---------------------------------------------------------------------------
// Main fused kernel: 4 waves/block, 16 pixels/wave, 16384 blocks.
// ---------------------------------------------------------------------------
__global__ __launch_bounds__(128)
void cnilut_main(const float* __restrict__ x, const float* __restrict__ W0,
                 const float* __restrict__ c0, const h16* __restrict__ wt,
                 const float* __restrict__ b1, const float* __restrict__ b2,
                 const float* __restrict__ b3, const float* __restrict__ w4t,
                 const float* __restrict__ b4, float* __restrict__ out) {
    __shared__ __align__(16) h16 act[WAVES * 16 * STRIDE];   // 33.8 KB
    __shared__ float rgbs[WAVES][16][3];

    const int lane = threadIdx.x & 31;
    const int w    = threadIdx.x >> 5;
    const int P0   = blockIdx.x * PIX_PER_WG + w * 16;       // 16-px tile never straddles images
    const int img  = P0 / HW;
    const int p0   = P0 - img * HW;
    const long base = (long)img * 3 * HW + p0;               // planar NCHW addressing

    h16* aw = &act[w * 16 * STRIDE];

    // ---- load this wave's 16 rgb pixels (planar, coalesced) ----
    if (lane < 16) {
        rgbs[w][lane][0] = x[base + lane];
        rgbs[w][lane][1] = x[base + HW + lane];
        rgbs[w][lane][2] = x[base + 2 * HW + lane];
    }
    __builtin_amdgcn_wave_barrier();   // same-wave LDS is in-order; just fence scheduling

    // ---- layer 0: h = relu(c0 + rgb @ W0[0:3]) -> f16 activation tile ----
    for (int idx = lane; idx < 16 * NF; idx += 32) {
        int m = idx >> 8, nn = idx & 255;
        float v = c0[nn] + rgbs[w][m][0] * W0[nn]
                         + rgbs[w][m][1] * W0[256 + nn]
                         + rgbs[w][m][2] * W0[512 + nn];
        aw[m * STRIDE + nn] = (h16)fmaxf(v, 0.0f);
    }
    __builtin_amdgcn_wave_barrier();

    // ---- 3 hidden layers: z = tanh(z @ W + b), all via v_wmma_f32_16x16x32_f16
    const int nrow  = lane & 15;          // N within tile (B/C/D lane mapping)
    const int koff  = (lane >> 4) * 8;    // B: lanes 16-31 hold K=8..15 / 24..31
    const int kbA   = (lane >> 4) * 8;    // A: lanes 16-31 hold K=8..15 / 24..31
    const int mbase = (lane >> 4) * 8;    // D: lanes 16-31 hold M=8..15

    #pragma unroll 1
    for (int layer = 0; layer < 3; ++layer) {
        // Load full A tile (16x256 f16 = 8 chunks of 16x32) into registers.
        v16h A[8];
        {
            int m = lane & 15;
            #pragma unroll
            for (int c = 0; c < 8; ++c) {
                v8h lo = *(v8h*)&aw[m * STRIDE + c * 32 + kbA];       // K = kbA..kbA+7
                v8h hi = *(v8h*)&aw[m * STRIDE + c * 32 + kbA + 16];  // K = kbA+16..+23
                v16h a;
                #pragma unroll
                for (int i = 0; i < 8; ++i) { a[i] = lo[i]; a[8 + i] = hi[i]; }
                A[c] = a;
            }
        }
        __builtin_amdgcn_wave_barrier();

        const h16*  Wl = wt + layer * 65536;
        const float* bl = (layer == 0) ? b1 : (layer == 1) ? b2 : b3;

        // Preload this lane's 16 bias values (SADDR-form global loads, one
        // clause at layer start; keeps FLAT/DScnt traffic out of the GEMM loop).
        float bc[16];
        #pragma unroll
        for (int t = 0; t < 16; ++t) bc[t] = bl[t * 16 + nrow];

        // Two independent accumulation chains per iteration (N-tiles nt, nt+8)
        // sharing the A registers -> 16 WMMAs per loop body, better XDL overlap.
        #pragma unroll 1
        for (int nt = 0; nt < 8; ++nt) {
            v8f acc0 = {}, acc1 = {};
            const h16* bp0 = Wl + (nt * 16 + nrow) * 256 + koff;        // N-tile nt
            const h16* bp1 = bp0 + 128 * 256;                           // N-tile nt+8
            #pragma unroll
            for (int kc = 0; kc < 8; ++kc) {
                v8h lo0 = *(const v8h*)(bp0 + kc * 32);
                v8h hi0 = *(const v8h*)(bp0 + kc * 32 + 16);
                v8h lo1 = *(const v8h*)(bp1 + kc * 32);
                v8h hi1 = *(const v8h*)(bp1 + kc * 32 + 16);
                v16h bv0, bv1;
                #pragma unroll
                for (int i = 0; i < 8; ++i) {
                    bv0[i] = lo0[i]; bv0[8 + i] = hi0[i];
                    bv1[i] = lo1[i]; bv1[8 + i] = hi1[i];
                }
                acc0 = __builtin_amdgcn_wmma_f32_16x16x32_f16(
                         false, A[kc], false, bv0, (short)0, acc0, false, false);
                acc1 = __builtin_amdgcn_wmma_f32_16x16x32_f16(
                         false, A[kc], false, bv1, (short)0, acc1, false, false);
            }
            // bias + tanh + pack f16 back into activation tile (D layout:
            // VGPR r = row mbase+r, col nt*16+nrow)
            float bn0 = bc[nt], bn1 = bc[nt + 8];
            #pragma unroll
            for (int r = 0; r < 8; ++r) {
                float v0 = fast_tanh(acc0[r] + bn0);
                float v1 = fast_tanh(acc1[r] + bn1);
                aw[(mbase + r) * STRIDE + nt * 16 + nrow]         = (h16)v0;
                aw[(mbase + r) * STRIDE + (nt + 8) * 16 + nrow]   = (h16)v1;
            }
        }
        __builtin_amdgcn_wave_barrier();
    }

    // ---- final 256->3 + residual + clip, planar coalesced store ----
    for (int idx = lane; idx < 48; idx += 32) {
        int m = idx / 3, c = idx - 3 * m;
        const float* wr = w4t + c * 256;
        float s = b4[c];
        #pragma unroll 4
        for (int k = 0; k < 256; k += 8) {
            v8h av = *(v8h*)&aw[m * STRIDE + k];
            #pragma unroll
            for (int i = 0; i < 8; ++i) s += (float)av[i] * wr[k + i];
        }
        float v = rgbs[w][m][c] + s;
        out[base + (long)c * HW + m] = fminf(fmaxf(v, 0.0f), 1.0f);
    }
}

// ---------------------------------------------------------------------------
extern "C" void kernel_launch(void* const* d_in, const int* in_sizes, int n_in,
                              void* d_out, int out_size, void* d_ws, size_t ws_size,
                              hipStream_t stream) {
    (void)in_sizes; (void)n_in; (void)out_size; (void)ws_size;
    const float* x     = (const float*)d_in[0];
    const float* style = (const float*)d_in[1];
    const float* W0    = (const float*)d_in[2];
    const float* b0    = (const float*)d_in[3];
    const float* W1    = (const float*)d_in[4];
    const float* b1    = (const float*)d_in[5];
    const float* W2    = (const float*)d_in[6];
    const float* b2    = (const float*)d_in[7];
    const float* W3    = (const float*)d_in[8];
    const float* b3    = (const float*)d_in[9];
    const float* W4    = (const float*)d_in[10];
    const float* b4    = (const float*)d_in[11];

    char*  ws  = (char*)d_ws;                 // needs ~388 KB of scratch
    h16*   wt  = (h16*)(ws + WS_WT);
    float* c0  = (float*)(ws + WS_C0);
    float* w4t = (float*)(ws + WS_W4T);

    int prep_work = 3 * 65536 + 256 + 768;
    cnilut_prep<<<(prep_work + 255) / 256, 256, 0, stream>>>(
        W0, b0, W1, W2, W3, W4, style, wt, c0, w4t);

    cnilut_main<<<NPIX / PIX_PER_WG, 128, 0, stream>>>(
        x, W0, c0, wt, b1, b2, b3, w4t, b4, (float*)d_out);
}
